// LongcatFlashMoE_29935922053179
// MI455X (gfx1250) — matrix operand order repaired and checked by reading, not compiled, not run on hardware
//
#include <hip/hip_runtime.h>
#include <cstdint>
#include <cstddef>

// ---------------- problem constants ----------------
static constexpr int TOK = 8192;   // tokens
static constexpr int H   = 1024;   // hidden
static constexpr int I   = 512;    // intermediate
static constexpr int E   = 8;      // routed experts
static constexpr int EZ  = 12;     // routed + zero experts
static constexpr float SCALE = 2.5f;

static constexpr int LDK = 40;     // padded LDS K-stride (elements) -> conflict-free b128 reads

// ---------------- vector types for WMMA ----------------
typedef __attribute__((ext_vector_type(16))) __bf16 v16bf;
typedef __attribute__((ext_vector_type(8)))  __bf16 v8bf;
typedef __attribute__((ext_vector_type(8)))  float  v8f;

__device__ __forceinline__ unsigned short f32_to_bf16(float f) {
    unsigned u = __builtin_bit_cast(unsigned, f);
    u += 0x7FFFu + ((u >> 16) & 1u);     // round-to-nearest-even
    return (unsigned short)(u >> 16);
}

__device__ __forceinline__ v16bf cat16(v8bf lo, v8bf hi) {
    return __builtin_shufflevector(lo, hi, 0,1,2,3,4,5,6,7,8,9,10,11,12,13,14,15);
}

// CDNA5 async global->LDS copy (ASYNCcnt-tracked); lds = wave-relative LDS byte
// offset in a VGPR, ga = 64-bit global address.
__device__ __forceinline__ void async_ld_b128(unsigned lds, unsigned long long ga) {
    asm volatile("global_load_async_to_lds_b128 %0, %1, off"
                 :: "v"(lds), "v"(ga) : "memory");
}
__device__ __forceinline__ void wait_async0() {
    asm volatile("s_wait_asynccnt 0x0" ::: "memory");
}

__device__ __forceinline__ unsigned lds_off(const void* p) {
    return (unsigned)(uintptr_t)p;   // generic LDS pointer: low 32 bits = LDS offset
}

// ---------------- fp32 -> bf16 conversion ----------------
__global__ void k_cvt_bf16(const float* __restrict__ src,
                           unsigned short* __restrict__ dst, int n) {
    int i = blockIdx.x * blockDim.x + threadIdx.x;
    int stride = gridDim.x * blockDim.x;
    for (; i < n; i += stride) dst[i] = f32_to_bf16(src[i]);
}

// ---------------- router: softmax + biased top-2 over 12 experts ----------------
__global__ void k_router(const float* __restrict__ x,
                         const float* __restrict__ rw,
                         const float* __restrict__ bias,
                         float* __restrict__ combine,
                         float* __restrict__ zscale) {
    const int lane = threadIdx.x & 31;
    const int wave = threadIdx.x >> 5;
    const int t    = blockIdx.x * 4 + wave;          // blockDim.x == 128
    if (t >= TOK) return;

    float s[EZ];
#pragma unroll
    for (int e = 0; e < EZ; ++e) s[e] = 0.0f;

    const float* xp = x + (size_t)t * H;
#pragma unroll 4
    for (int j = 0; j < H / 32; ++j) {
        float xv = xp[j * 32 + lane];
#pragma unroll
        for (int e = 0; e < EZ; ++e)
            s[e] += xv * rw[e * H + j * 32 + lane];
    }
#pragma unroll
    for (int e = 0; e < EZ; ++e) {
#pragma unroll
        for (int off = 16; off > 0; off >>= 1)
            s[e] += __shfl_xor(s[e], off, 32);
    }

    float m = s[0];
#pragma unroll
    for (int e = 1; e < EZ; ++e) m = fmaxf(m, s[e]);
    float p[EZ], sum = 0.0f;
#pragma unroll
    for (int e = 0; e < EZ; ++e) { p[e] = __expf(s[e] - m); sum += p[e]; }
    float inv = __builtin_amdgcn_rcpf(sum);
    float sel[EZ];
#pragma unroll
    for (int e = 0; e < EZ; ++e) { p[e] *= inv; sel[e] = p[e] + bias[e]; }

    int i1 = 0;
#pragma unroll
    for (int e = 1; e < EZ; ++e) if (sel[e] > sel[i1]) i1 = e;
    float w1 = p[i1];
    sel[i1] = -1e30f;
    int i2 = 0;
#pragma unroll
    for (int e = 1; e < EZ; ++e) if (sel[e] > sel[i2]) i2 = e;
    float w2 = p[i2];

    float cw[E];
#pragma unroll
    for (int e = 0; e < E; ++e) cw[e] = 0.0f;
    float z = 0.0f;
    if (i1 >= E) z += w1; else cw[i1] = w1 * SCALE;
    if (i2 >= E) z += w2; else cw[i2] = w2 * SCALE;

    if (lane == 0) {
#pragma unroll
        for (int e = 0; e < E; ++e) combine[(size_t)t * E + e] = cw[e];
        zscale[t] = z;
    }
}

// ---------------- init: out = zero_scale[t] * x ----------------
__global__ void k_init_out(const float* __restrict__ x,
                           const float* __restrict__ zscale,
                           float* __restrict__ out) {
    int i = blockIdx.x * blockDim.x + threadIdx.x;
    int stride = gridDim.x * blockDim.x;
    for (; i < TOK * H; i += stride)
        out[i] = zscale[i >> 10] * x[i];       // H == 1024
}

// ---------------- GEMM1: act = silu(x@Wg^T) * (x@Wu^T) * combine ----------------
// 128x64 block tile, K-step 32 over H, double-buffered LDS filled by async copies.
__global__ __launch_bounds__(256) void k_gemm1(
        const unsigned short* __restrict__ xb,     // [TOK,H] bf16
        const unsigned short* __restrict__ w13b,   // [E,2I,H] bf16
        const float* __restrict__ combine,         // [TOK,E] (already *SCALE)
        unsigned short* __restrict__ actb,         // [TOK,I] bf16
        int e) {
    __shared__ __align__(16) unsigned short As[2][128 * LDK];
    __shared__ __align__(16) unsigned short Bg[2][64 * LDK];
    __shared__ __align__(16) unsigned short Bu[2][64 * LDK];
    constexpr unsigned ASZ = 128 * LDK * 2, BSZ = 64 * LDK * 2;

    const int tid  = threadIdx.x;
    const int lane = tid & 31, wave = tid >> 5;
    const int wm = wave >> 1, wn = wave & 1;
    const int l16 = lane & 15, half = lane >> 4;
    const int m0 = blockIdx.x * 128;
    const int n0 = blockIdx.y * 64;
    const unsigned short* wp = w13b + (size_t)e * (2 * I * H);

    // per-thread 16B copy slots
    const int row = tid >> 2;            // 0..63
    const int kc  = (tid & 3) * 8;       // 0,8,16,24
    const unsigned long long gA0 = (unsigned long long)(uintptr_t)
        &xb[(size_t)(m0 + row) * H + kc];
    const unsigned long long gA1 = (unsigned long long)(uintptr_t)
        &xb[(size_t)(m0 + row + 64) * H + kc];
    const unsigned long long gBg = (unsigned long long)(uintptr_t)
        &wp[(size_t)(n0 + row) * H + kc];
    const unsigned long long gBu = (unsigned long long)(uintptr_t)
        &wp[(size_t)(I + n0 + row) * H + kc];
    const unsigned lA0 = lds_off(&As[0][row * LDK + kc]);
    const unsigned lA1 = lds_off(&As[0][(row + 64) * LDK + kc]);
    const unsigned lBg = lds_off(&Bg[0][row * LDK + kc]);
    const unsigned lBu = lds_off(&Bu[0][row * LDK + kc]);

    auto issue = [&](int kt, int buf) {
        unsigned long long go = (unsigned long long)kt * 64;  // 32 bf16 per K-step
        async_ld_b128(lA0 + buf * ASZ, gA0 + go);
        async_ld_b128(lA1 + buf * ASZ, gA1 + go);
        async_ld_b128(lBg + buf * BSZ, gBg + go);
        async_ld_b128(lBu + buf * BSZ, gBu + go);
    };

    v8f ag[2][2], au[2][2];
#pragma unroll
    for (int mi = 0; mi < 2; ++mi)
#pragma unroll
        for (int ni = 0; ni < 2; ++ni) { ag[mi][ni] = (v8f){}; au[mi][ni] = (v8f){}; }

    constexpr int NK = H / 32;
    issue(0, 0);
    for (int k = 0; k < NK; ++k) {
        const int cur = k & 1;
        wait_async0();        // my async fills of buffer `cur` are in LDS
        __syncthreads();      // everyone's fills done; everyone done reading cur^1
        const int kn = (k + 1 < NK) ? k + 1 : k;   // redundant refetch on last iter
        issue(kn, cur ^ 1);   // overlaps with compute below

        v16bf a[2], bgf[2], buf_[2];
#pragma unroll
        for (int mi = 0; mi < 2; ++mi) {
            int r = wm * 32 + mi * 16 + l16;
            a[mi] = cat16(*(const v8bf*)&As[cur][r * LDK + half * 8],
                          *(const v8bf*)&As[cur][r * LDK + 16 + half * 8]);
        }
#pragma unroll
        for (int ni = 0; ni < 2; ++ni) {
            int c  = wn * 32 + ni * 16 + l16;
            int ko = half * 8;
            bgf[ni]  = cat16(*(const v8bf*)&Bg[cur][c * LDK + ko],
                             *(const v8bf*)&Bg[cur][c * LDK + ko + 16]);
            buf_[ni] = cat16(*(const v8bf*)&Bu[cur][c * LDK + ko],
                             *(const v8bf*)&Bu[cur][c * LDK + ko + 16]);
        }
#pragma unroll
        for (int mi = 0; mi < 2; ++mi)
#pragma unroll
            for (int ni = 0; ni < 2; ++ni) {
                ag[mi][ni] = __builtin_amdgcn_wmma_f32_16x16x32_bf16(
                    false, a[mi], false, bgf[ni], (short)0, ag[mi][ni], false, false);
                au[mi][ni] = __builtin_amdgcn_wmma_f32_16x16x32_bf16(
                    false, a[mi], false, buf_[ni], (short)0, au[mi][ni], false, false);
            }
    }

    // epilogue: silu(g)*u * combine -> bf16   (v_rcp_f32 instead of IEEE divide)
#pragma unroll
    for (int mi = 0; mi < 2; ++mi)
#pragma unroll
        for (int ni = 0; ni < 2; ++ni)
#pragma unroll
            for (int r = 0; r < 8; ++r) {
                int t = m0 + wm * 32 + mi * 16 + half * 8 + r;
                int n = n0 + wn * 32 + ni * 16 + l16;
                float g = ag[mi][ni][r];
                float u = au[mi][ni][r];
                float cwv = combine[(size_t)t * E + e];
                float sig = __builtin_amdgcn_rcpf(1.0f + __expf(-g));
                actb[(size_t)t * I + n] = f32_to_bf16(g * sig * u * cwv);
            }
}

// ---------------- GEMM2: out += act @ W2^T (combine already folded into act) --
__global__ __launch_bounds__(256) void k_gemm2(
        const unsigned short* __restrict__ actb,   // [TOK,I] bf16
        const unsigned short* __restrict__ w2b,    // [E,H,I] bf16
        float* __restrict__ out,                   // [TOK,H] fp32 accumulate
        int e) {
    __shared__ __align__(16) unsigned short As[2][128 * LDK];
    __shared__ __align__(16) unsigned short Bs[2][64 * LDK];
    constexpr unsigned ASZ = 128 * LDK * 2, BSZ = 64 * LDK * 2;

    const int tid  = threadIdx.x;
    const int lane = tid & 31, wave = tid >> 5;
    const int wm = wave >> 1, wn = wave & 1;
    const int l16 = lane & 15, half = lane >> 4;
    const int m0 = blockIdx.x * 128;
    const int n0 = blockIdx.y * 64;
    const unsigned short* wp = w2b + (size_t)e * (H * I);

    const int row = tid >> 2;
    const int kc  = (tid & 3) * 8;
    const unsigned long long gA0 = (unsigned long long)(uintptr_t)
        &actb[(size_t)(m0 + row) * I + kc];
    const unsigned long long gA1 = (unsigned long long)(uintptr_t)
        &actb[(size_t)(m0 + row + 64) * I + kc];
    const unsigned long long gB  = (unsigned long long)(uintptr_t)
        &wp[(size_t)(n0 + row) * I + kc];
    const unsigned lA0 = lds_off(&As[0][row * LDK + kc]);
    const unsigned lA1 = lds_off(&As[0][(row + 64) * LDK + kc]);
    const unsigned lB  = lds_off(&Bs[0][row * LDK + kc]);

    auto issue = [&](int kt, int buf) {
        unsigned long long go = (unsigned long long)kt * 64;
        async_ld_b128(lA0 + buf * ASZ, gA0 + go);
        async_ld_b128(lA1 + buf * ASZ, gA1 + go);
        async_ld_b128(lB  + buf * BSZ, gB  + go);
    };

    v8f acc[2][2];
#pragma unroll
    for (int mi = 0; mi < 2; ++mi)
#pragma unroll
        for (int ni = 0; ni < 2; ++ni) acc[mi][ni] = (v8f){};

    constexpr int NK = I / 32;
    issue(0, 0);
    for (int k = 0; k < NK; ++k) {
        const int cur = k & 1;
        wait_async0();
        __syncthreads();
        const int kn = (k + 1 < NK) ? k + 1 : k;
        issue(kn, cur ^ 1);

        v16bf a[2], b[2];
#pragma unroll
        for (int mi = 0; mi < 2; ++mi) {
            int r = wm * 32 + mi * 16 + l16;
            a[mi] = cat16(*(const v8bf*)&As[cur][r * LDK + half * 8],
                          *(const v8bf*)&As[cur][r * LDK + 16 + half * 8]);
        }
#pragma unroll
        for (int ni = 0; ni < 2; ++ni) {
            int c  = wn * 32 + ni * 16 + l16;
            int ko = half * 8;
            b[ni] = cat16(*(const v8bf*)&Bs[cur][c * LDK + ko],
                          *(const v8bf*)&Bs[cur][c * LDK + ko + 16]);
        }
#pragma unroll
        for (int mi = 0; mi < 2; ++mi)
#pragma unroll
            for (int ni = 0; ni < 2; ++ni)
                acc[mi][ni] = __builtin_amdgcn_wmma_f32_16x16x32_bf16(
                    false, a[mi], false, b[ni], (short)0, acc[mi][ni], false, false);
    }

#pragma unroll
    for (int mi = 0; mi < 2; ++mi)
#pragma unroll
        for (int ni = 0; ni < 2; ++ni)
#pragma unroll
            for (int r = 0; r < 8; ++r) {
                int t = m0 + wm * 32 + mi * 16 + half * 8 + r;
                int n = n0 + wn * 32 + ni * 16 + l16;
                out[(size_t)t * H + n] += acc[mi][ni][r];
            }
}

// ---------------- launcher ----------------
extern "C" void kernel_launch(void* const* d_in, const int* in_sizes, int n_in,
                              void* d_out, int out_size, void* d_ws, size_t ws_size,
                              hipStream_t stream) {
    const float* x    = (const float*)d_in[0];   // [TOK,H]
    const float* rw   = (const float*)d_in[1];   // [EZ,H]
    const float* bias = (const float*)d_in[2];   // [EZ]
    const float* w13  = (const float*)d_in[3];   // [E,2I,H]
    const float* w2   = (const float*)d_in[4];   // [E,H,I]
    float* out = (float*)d_out;                  // [TOK,H]

    char* w = (char*)d_ws;
    unsigned short* xb   = (unsigned short*)w;                 w += (size_t)TOK * H * 2;
    unsigned short* w13b = (unsigned short*)w;                 w += (size_t)E * 2 * I * H * 2;
    unsigned short* w2b  = (unsigned short*)w;                 w += (size_t)E * H * I * 2;
    unsigned short* actb = (unsigned short*)w;                 w += (size_t)TOK * I * 2;
    float* combine       = (float*)w;                          w += (size_t)TOK * E * 4;
    float* zscale        = (float*)w;

    k_cvt_bf16<<<4096, 256, 0, stream>>>(x,   xb,   TOK * H);
    k_cvt_bf16<<<4096, 256, 0, stream>>>(w13, w13b, E * 2 * I * H);
    k_cvt_bf16<<<4096, 256, 0, stream>>>(w2,  w2b,  E * H * I);

    k_router<<<TOK / 4, 128, 0, stream>>>(x, rw, bias, combine, zscale);
    k_init_out<<<4096, 256, 0, stream>>>(x, zscale, out);

    for (int e = 0; e < E; ++e) {
        k_gemm1<<<dim3(TOK / 128, I / 64),  256, 0, stream>>>(xb, w13b, combine, actb, e);
        k_gemm2<<<dim3(TOK / 128, H / 64), 256, 0, stream>>>(actb, w2b, out, e);
    }
}
